// OneHot_10582799418087
// MI455X (gfx1250) — matrix operand order repaired and checked by reading, not compiled, not run on hardware
//
#include <hip/hip_runtime.h>

typedef __attribute__((ext_vector_type(16))) _Float16 v16h;
typedef __attribute__((ext_vector_type(8)))  float    v8f;
typedef __attribute__((ext_vector_type(4)))  float    v4f;

#define N_FEAT 64        // K dimension
#define N_COLS 512       // N dimension (64 features * 8 buckets)
#define NTILES 32        // N_COLS / 16
#define ROWS_PER_BLOCK 128

// D = A*B + C with A = x tile (16x64 f16), B = m tile (64x16 f16), fused epilogue.
__global__ __launch_bounds__(256) void onehot_wmma_kernel(
    const float* __restrict__ x,     // [rows, 64]
    const float* __restrict__ m,     // [64, 512]
    const float* __restrict__ bias,  // [512]
    const float* __restrict__ caps,  // [512]
    const float* __restrict__ scale, // [512]
    float* __restrict__ out)         // [rows, 512]
{
    // 64 B-fragments (32 tiles x 2 K-halves), each 32 lanes * 16 halves = 1KB -> 64KB
    __shared__ __attribute__((aligned(32))) _Float16 bimg[NTILES * 2 * 512];
    // Packed epilogue params: [n] -> {bias, caps, scale, pad}  (8KB)
    __shared__ __attribute__((aligned(16))) float epil[N_COLS * 4];

    const int tid = threadIdx.x;

    // ---- Phase 0a: build B fragments in LDS (cooperative, no divergence) ----
    // B-fragment layout (wave32, 16-bit B 32x16): lane L holds column n = L%16,
    // element v (v=0..15, packed 2/VGPR) holds K = 32*h + 8*(L>=16) + v + 8*(v>=8).
    for (int t = tid; t < 64 * 256; t += 256) {
        const int fragIdx = t >> 8;        // 0..63  (= tile*2 + h)
        const int rem     = t & 255;
        const int slot    = rem >> 3;      // lane slot 0..31
        const int d       = rem & 7;       // dword 0..7 (two f16 each)
        const int ntile   = fragIdx >> 1;
        const int h       = fragIdx & 1;   // K-half (0: K 0..31, 1: K 32..63)
        const int q       = slot >> 4;     // lane half
        const int n       = ntile * 16 + (slot & 15);
        const int k0      = 32 * h + 8 * q + 2 * d + ((d >> 2) << 3);
        const _Float16 lo = (_Float16)m[(size_t)k0 * N_COLS + n];
        const _Float16 hi = (_Float16)m[(size_t)(k0 + 1) * N_COLS + n];
        const int idx = fragIdx * 512 + slot * 16 + 2 * d;
        bimg[idx]     = lo;
        bimg[idx + 1] = hi;
    }
    // ---- Phase 0b: stage epilogue params (keeps steady loop free of VMEM loads)
    for (int n = tid; n < N_COLS; n += 256) {
        epil[n * 4 + 0] = bias[n];
        epil[n * 4 + 1] = caps[n];
        epil[n * 4 + 2] = scale[n];
        epil[n * 4 + 3] = 0.0f;
    }
    __syncthreads();

    // ---- Phase 1: per-wave 16-row strip GEMM with fused epilogue ----
    const int lane = tid & 31;
    const int wid  = tid >> 5;            // 8 waves/block
    const int q    = lane >> 4;           // lane half
    const int nl   = lane & 15;
    const size_t rowBase = (size_t)blockIdx.x * ROWS_PER_BLOCK + (size_t)wid * 16;

    // A-fragment (16-bit A 16x32 ISA layout): lane holds row M = lane%16;
    // lanes<16: K = {0..7, 16..23}; lanes>=16: K = {8..15, 24..31}; +32 for frag 1.
    const float* xr = x + (rowBase + (size_t)nl) * N_FEAT;
    const int kq = q * 8;
    v16h a0, a1;
    {
        const v4f c0 = *(const v4f*)(xr + kq);
        const v4f c1 = *(const v4f*)(xr + kq + 4);
        const v4f c2 = *(const v4f*)(xr + kq + 16);
        const v4f c3 = *(const v4f*)(xr + kq + 20);
        const v4f c4 = *(const v4f*)(xr + kq + 32);
        const v4f c5 = *(const v4f*)(xr + kq + 36);
        const v4f c6 = *(const v4f*)(xr + kq + 48);
        const v4f c7 = *(const v4f*)(xr + kq + 52);
#pragma unroll
        for (int i = 0; i < 4; ++i) {
            a0[i]      = (_Float16)c0[i];
            a0[4 + i]  = (_Float16)c1[i];
            a0[8 + i]  = (_Float16)c2[i];
            a0[12 + i] = (_Float16)c3[i];
            a1[i]      = (_Float16)c4[i];
            a1[4 + i]  = (_Float16)c5[i];
            a1[8 + i]  = (_Float16)c6[i];
            a1[12 + i] = (_Float16)c7[i];
        }
    }

    // C/D layout: VGPR v, lanes 0-15 -> M=v, N=lane; lanes 16-31 -> M=8+v, N=lane-16.
    float* outp = out + (rowBase + (size_t)(8 * q)) * N_COLS + nl;

#pragma unroll 2
    for (int t = 0; t < NTILES; ++t) {
        const v16h b0 = *(const v16h*)&bimg[(2 * t + 0) * 512 + lane * 16];
        const v16h b1 = *(const v16h*)&bimg[(2 * t + 1) * 512 + lane * 16];
        v8f acc = {};
        acc = __builtin_amdgcn_wmma_f32_16x16x32_f16(false, a0, false, b0,
                                                     (short)0, acc, false, false);
        acc = __builtin_amdgcn_wmma_f32_16x16x32_f16(false, a1, false, b1,
                                                     (short)0, acc, false, false);

        const int n  = t * 16 + nl;        // all 8 accumulators share column n
        const v4f ep = *(const v4f*)&epil[n * 4];  // one ds_load_b128, conflict-free
        const float bn = ep[0];
        const float cn = ep[1];
        const float sn = ep[2];
#pragma unroll
        for (int v = 0; v < 8; ++v) {
            float y = acc[v] + bn;
            y = fmaxf(y, 0.0f);
            y = (y < cn) ? y : 0.0f;
            y *= sn;
            // streaming output: write-once, bypass caches (non-temporal)
            __builtin_nontemporal_store(y, outp + (size_t)v * N_COLS + t * 16);
        }
    }
}

extern "C" void kernel_launch(void* const* d_in, const int* in_sizes, int n_in,
                              void* d_out, int out_size, void* d_ws, size_t ws_size,
                              hipStream_t stream) {
    const float* x     = (const float*)d_in[0];
    const float* m     = (const float*)d_in[1];
    const float* bias  = (const float*)d_in[2];
    const float* caps  = (const float*)d_in[3];
    const float* scale = (const float*)d_in[4];
    float* out = (float*)d_out;

    const int rows = in_sizes[0] / N_FEAT;        // 262144
    const int grid = rows / ROWS_PER_BLOCK;       // 2048 blocks of 256 (8 waves)
    onehot_wmma_kernel<<<grid, 256, 0, stream>>>(x, m, bias, caps, scale, out);
}
